// _HeteroLayer_66709432042192
// MI455X (gfx1250) — matrix operand order repaired and checked by reading, not compiled, not run on hardware
//
#include <hip/hip_runtime.h>
#include <math.h>

typedef __attribute__((ext_vector_type(16))) _Float16 v16h;
typedef __attribute__((ext_vector_type(8)))  float    v8f;
typedef __attribute__((ext_vector_type(4)))  _Float16 hv4;
typedef __attribute__((ext_vector_type(4)))  int      v4i;

typedef __attribute__((address_space(1))) v4i v4i_g;   // global (device) int32x4
typedef __attribute__((address_space(3))) v4i v4i_l;   // LDS int32x4

#define HD 128
#define FFND 512

// ---------------- CDNA5 async global->LDS staging (compile-safe fallback) ----------------
#if defined(__has_builtin)
#if __has_builtin(__builtin_amdgcn_global_load_async_to_lds_b128)
#define HAVE_ASYNC_LDS 1
#endif
#endif

__device__ __forceinline__ void copy16_to_lds(const _Float16* gsrc, _Float16* ldst) {
#ifdef HAVE_ASYNC_LDS
    __builtin_amdgcn_global_load_async_to_lds_b128(
        (v4i_g*)(void*)const_cast<_Float16*>(gsrc), (v4i_l*)(void*)ldst, 0, 0);
#else
    *reinterpret_cast<uint4*>(ldst) = *reinterpret_cast<const uint4*>(gsrc);
#endif
}
__device__ __forceinline__ void async_join() {
#ifdef HAVE_ASYNC_LDS
#if __has_builtin(__builtin_amdgcn_s_wait_asynccnt)
    __builtin_amdgcn_s_wait_asynccnt(0);
#else
    asm volatile("s_wait_asynccnt 0" ::: "memory");
#endif
#endif
}

// ---------------- helpers ----------------
__device__ __forceinline__ float gelu_exact(float x) {
    return 0.5f * x * (1.0f + erff(x * 0.70710678118654752440f));
}
__device__ __forceinline__ unsigned f2key(float f) {
    unsigned u = __float_as_uint(f);
    return (u & 0x80000000u) ? ~u : (u | 0x80000000u);
}
__device__ __forceinline__ float key2f(unsigned u) {
    return (u & 0x80000000u) ? __uint_as_float(u ^ 0x80000000u) : __uint_as_float(~u);
}

// ---------------- LayerNorm (fp32 in -> fp16 out), one wave per row ----------------
__global__ __launch_bounds__(256)
void ln_to_f16(const float* __restrict__ h, const float* __restrict__ g,
               const float* __restrict__ b, _Float16* __restrict__ out, int N) {
    const int wave = threadIdx.x >> 5, lane = threadIdx.x & 31;
    const int row = blockIdx.x * 8 + wave;
    if (row >= N) return;
    const float4 x = *reinterpret_cast<const float4*>(h + (size_t)row * HD + lane * 4);
    float s = x.x + x.y + x.z + x.w;
#pragma unroll
    for (int o = 16; o > 0; o >>= 1) s += __shfl_xor(s, o, 32);
    const float mean = s * (1.0f / 128.0f);
    const float d0 = x.x - mean, d1 = x.y - mean, d2 = x.z - mean, d3 = x.w - mean;
    float v = d0 * d0 + d1 * d1 + d2 * d2 + d3 * d3;
#pragma unroll
    for (int o = 16; o > 0; o >>= 1) v += __shfl_xor(v, o, 32);
    const float rs = rsqrtf(v * (1.0f / 128.0f) + 1e-5f);
    const float4 gg = *reinterpret_cast<const float4*>(g + lane * 4);
    const float4 bb = *reinterpret_cast<const float4*>(b + lane * 4);
    hv4 o;
    o[0] = (_Float16)(d0 * rs * gg.x + bb.x);
    o[1] = (_Float16)(d1 * rs * gg.y + bb.y);
    o[2] = (_Float16)(d2 * rs * gg.z + bb.z);
    o[3] = (_Float16)(d3 * rs * gg.w + bb.w);
    *reinterpret_cast<hv4*>(out + (size_t)row * HD + lane * 4) = o;
}

// ---------------- fp32 -> fp16 convert ----------------
__global__ void f32_to_f16(const float* __restrict__ in, _Float16* __restrict__ out, int n) {
    int i = blockIdx.x * blockDim.x + threadIdx.x;
    if (i < n) out[i] = (_Float16)in[i];
}

// ---------------- precompute u[r][sd][k][h] = sum_d wl[r][k][h*32+d] * a[r][h][d] ----------------
__global__ void prep_u(const float* __restrict__ wl, const float* __restrict__ a_src,
                       const float* __restrict__ a_dst, float* __restrict__ u) {
    int i = blockIdx.x * blockDim.x + threadIdx.x;
    if (i >= 5 * 2 * 128 * 4) return;
    int h = i & 3, k = (i >> 2) & 127, sd = (i >> 9) & 1, r = i >> 10;
    const float* a = (sd ? a_dst : a_src) + r * 128 + h * 32;
    const float* w = wl + r * 16384 + k * 128 + h * 32;
    float s = 0.f;
#pragma unroll
    for (int d = 0; d < 32; ++d) s += w[d] * a[d];
    u[i] = s;
}

// ---------------- precompute ce[r][k][h] = sum_d we[r][k][h*32+d] * a_edge[r][h][d] ----------------
__global__ void prep_ce(const float* __restrict__ we, const float* __restrict__ ae,
                        float* __restrict__ ce) {
    int i = threadIdx.x;
    if (i >= 128) return;
    int h = i & 3, k = (i >> 2) & 7, r = i >> 5;
    const float* w = we + r * 1024 + k * 128 + h * 32;
    const float* a = ae + r * 128 + h * 32;
    float s = 0.f;
#pragma unroll
    for (int d = 0; d < 32; ++d) s += w[d] * a[d];
    ce[i] = s;
}

// ---------------- node logit: one thread per node, all 4 heads ----------------
__global__ void node_dot(const _Float16* __restrict__ x, const float* __restrict__ u,
                         float* __restrict__ out, int N) {
    int n = blockIdx.x * blockDim.x + threadIdx.x;
    if (n >= N) return;
    const _Float16* xrow = x + (size_t)n * HD;
    float s0 = 0.f, s1 = 0.f, s2 = 0.f, s3 = 0.f;
#pragma unroll 4
    for (int k = 0; k < 128; ++k) {
        const float xv = (float)xrow[k];
        s0 += xv * u[k * 4 + 0];
        s1 += xv * u[k * 4 + 1];
        s2 += xv * u[k * 4 + 2];
        s3 += xv * u[k * 4 + 3];
    }
    *reinterpret_cast<float4*>(out + (size_t)n * 4) = make_float4(s0, s1, s2, s3);
}

// ---------------- WMMA GEMM: C[M,N] = A[M,K](f16) x B[K,N](f16) + bias ----------------
// block = 256 thr (8 waves); each wave -> 16 rows; block covers 128 rows x 128 cols.
// B slab double-buffered in LDS (async-to-LDS); per k-step all 8 B fragments are
// preloaded into independent registers so LDS latency overlaps the WMMA stream.
// EPI: 0 = f32 out, 1 = fp16 GELU out, 2 = f32 out + residual R, 3 = fp16 out
template <int EPI>
__global__ __launch_bounds__(256)
void gemm_wmma(const _Float16* __restrict__ A, int lda,
               const _Float16* __restrict__ B, int ldb,
               const float* __restrict__ bias,
               float* __restrict__ Cf, _Float16* __restrict__ Ch,
               const float* __restrict__ R, int ldc, int K) {
    __shared__ _Float16 Bs[2][128 * 128];
    const int tid = threadIdx.x;
    const int wave = tid >> 5, lane = tid & 31;
    const int row0 = (blockIdx.x * 8 + wave) * 16;
    const int col0 = blockIdx.y * 128;
    const int hi = lane >> 4;

    const int r = tid >> 1;                 // staging: 2 threads per B row
    const int cH = (tid & 1) * 64;          // 64 halves (128B) per thread

    v8f acc[8] = {};

    // stage chunk 0
    {
        const _Float16* src = B + (size_t)r * ldb + col0 + cH;
        _Float16* dst = &Bs[0][r * 128 + cH];
#pragma unroll
        for (int i = 0; i < 8; ++i) copy16_to_lds(src + i * 8, dst + i * 8);
    }
    async_join();
    __syncthreads();

    const int nChunks = K >> 7;
    for (int c = 0; c < nChunks; ++c) {
        const int buf = c & 1;
        if (c + 1 < nChunks) {  // prefetch next B slab into other LDS buffer
            const _Float16* src = B + (size_t)((c + 1) * 128 + r) * ldb + col0 + cH;
            _Float16* dst = &Bs[buf ^ 1][r * 128 + cH];
#pragma unroll
            for (int i = 0; i < 8; ++i) copy16_to_lds(src + i * 8, dst + i * 8);
        }

        const _Float16* aRowP = A + (size_t)(row0 + (lane & 15)) * lda + c * 128;
        const _Float16* bBase = &Bs[buf][0];
#pragma unroll
        for (int kk = 0; kk < 128; kk += 32) {
            // A fragment: lane (m = lane&15); halves 0-7 <- K=kk+hi*8.., 8-15 <- K=kk+16+hi*8..
            union { uint4 u[2]; v16h v; } au;
            au.u[0] = *reinterpret_cast<const uint4*>(aRowP + kk + hi * 8);
            au.u[1] = *reinterpret_cast<const uint4*>(aRowP + kk + 16 + hi * 8);
            // B fragments: lane = N column (lane&15); halves 0-7 <- K=kk+hi*8.., 8-15 <- K=kk+16+hi*8..
            // Load all 8 tiles into independent registers, then issue 8 WMMAs.
            const _Float16* bCol = bBase + (kk + hi * 8) * 128 + (lane & 15);
            union { uint4 u[2]; v16h v; } bf[8];
#pragma unroll
            for (int t = 0; t < 8; ++t) {
#pragma unroll
                for (int s = 0; s < 16; ++s)
                    bf[t].v[s] = bCol[((s & 7) + (s >> 3) * 16) * 128 + t * 16];
            }
#pragma unroll
            for (int t = 0; t < 8; ++t) {
                acc[t] = __builtin_amdgcn_wmma_f32_16x16x32_f16(
                    false, au.v, false, bf[t].v, (short)0, acc[t], false, false);
            }
        }
        async_join();
        __syncthreads();
    }

    // epilogue: C lane layout — lanes 0-15: N=lane, M=g; lanes 16-31: N=lane-16, M=8+g
    const int rBase = row0 + hi * 8;
    const int cBase = col0 + (lane & 15);
#pragma unroll
    for (int t = 0; t < 8; ++t) {
        const int c = cBase + t * 16;
        const float bv = bias ? bias[c] : 0.f;
#pragma unroll
        for (int g = 0; g < 8; ++g) {
            const size_t idx = (size_t)(rBase + g) * ldc + c;
            float val = acc[t][g] + bv;
            if (EPI == 0)      Cf[idx] = val;
            else if (EPI == 1) Ch[idx] = (_Float16)gelu_exact(val);
            else if (EPI == 2) Cf[idx] = R[idx] + val;
            else               Ch[idx] = (_Float16)val;
        }
    }
}

// ---------------- phys: scatter msg[src] into sum[dst], count deg ----------------
__global__ __launch_bounds__(256)
void phys_scatter(const _Float16* __restrict__ P, const int* __restrict__ src,
                  const int* __restrict__ dst, float* __restrict__ sum,
                  float* __restrict__ deg, int E) {
    const long gid = (long)blockIdx.x * blockDim.x + threadIdx.x;
    const int e = (int)(gid >> 5);
    if (e >= E) return;
    const int lane = (int)(gid & 31);
    const int s = src[e], d = dst[e];
    const _Float16* p = P + (size_t)s * HD + lane * 4;
    float* o = sum + (size_t)d * HD + lane * 4;
    atomicAdd(o + 0, (float)p[0]);
    atomicAdd(o + 1, (float)p[1]);
    atomicAdd(o + 2, (float)p[2]);
    atomicAdd(o + 3, (float)p[3]);
    if (lane == 0) atomicAdd(deg + d, 1.0f);
}

// ---------------- GAT pass 1: logits + segment max ----------------
__global__ __launch_bounds__(256)
void gat_logit(const float* __restrict__ ls, const float* __restrict__ ld,
               const float* __restrict__ ef, const float* __restrict__ ce,
               const int* __restrict__ src, const int* __restrict__ dst,
               float* __restrict__ logit, unsigned* __restrict__ mx, int E) {
    int e = blockIdx.x * blockDim.x + threadIdx.x;
    if (e >= E) return;
    const int s = src[e], d = dst[e];
    const float4 a = *reinterpret_cast<const float4*>(ls + 4 * (size_t)s);
    const float4 b = *reinterpret_cast<const float4*>(ld + 4 * (size_t)d);
    float lg[4] = {a.x + b.x, a.y + b.y, a.z + b.z, a.w + b.w};
    if (ef) {
        const float* efe = ef + (size_t)e * 8;
#pragma unroll
        for (int k = 0; k < 8; ++k) {
            const float f = efe[k];
#pragma unroll
            for (int h = 0; h < 4; ++h) lg[h] += f * ce[k * 4 + h];
        }
    }
#pragma unroll
    for (int h = 0; h < 4; ++h) {
        lg[h] = lg[h] > 0.f ? lg[h] : 0.2f * lg[h];
        atomicMax(mx + (size_t)d * 4 + h, f2key(lg[h]));
    }
    *reinterpret_cast<float4*>(logit + (size_t)e * 4) = make_float4(lg[0], lg[1], lg[2], lg[3]);
}

// ---------------- GAT pass 2: exp + segment sum ----------------
__global__ __launch_bounds__(256)
void gat_exp(float* __restrict__ logit_ex, const unsigned* __restrict__ mx,
             float* __restrict__ den, const int* __restrict__ dst, int E) {
    int e = blockIdx.x * blockDim.x + threadIdx.x;
    if (e >= E) return;
    const int d = dst[e];
    float4 lg = *reinterpret_cast<float4*>(logit_ex + (size_t)e * 4);
    float ex[4];
    ex[0] = expf(lg.x - key2f(mx[(size_t)d * 4 + 0]));
    ex[1] = expf(lg.y - key2f(mx[(size_t)d * 4 + 1]));
    ex[2] = expf(lg.z - key2f(mx[(size_t)d * 4 + 2]));
    ex[3] = expf(lg.w - key2f(mx[(size_t)d * 4 + 3]));
#pragma unroll
    for (int h = 0; h < 4; ++h) atomicAdd(den + (size_t)d * 4 + h, ex[h]);
    *reinterpret_cast<float4*>(logit_ex + (size_t)e * 4) = make_float4(ex[0], ex[1], ex[2], ex[3]);
}

// ---------------- GAT pass 3: alpha * z[src] scatter ----------------
__global__ __launch_bounds__(256)
void gat_scatter(const _Float16* __restrict__ Z, const float* __restrict__ ex,
                 const float* __restrict__ den, const int* __restrict__ src,
                 const int* __restrict__ dst, float* __restrict__ acc, int E) {
    const long gid = (long)blockIdx.x * blockDim.x + threadIdx.x;
    const int e = (int)(gid >> 5);
    if (e >= E) return;
    const int lane = (int)(gid & 31);
    const int s = src[e], d = dst[e];
    const int h = lane >> 3;  // 4 cols/lane => head = col/32 = lane/8
    const float al = ex[(size_t)e * 4 + h] / fmaxf(den[(size_t)d * 4 + h], 1e-9f);
    const _Float16* z = Z + (size_t)s * HD + lane * 4;
    float* o = acc + (size_t)d * HD + lane * 4;
    atomicAdd(o + 0, al * (float)z[0]);
    atomicAdd(o + 1, al * (float)z[1]);
    atomicAdd(o + 2, al * (float)z[2]);
    atomicAdd(o + 3, al * (float)z[3]);
}

// ---------------- residual combines ----------------
__global__ void combine_e(const float* __restrict__ h0, const float* __restrict__ gacc,
                          float* __restrict__ out, long n) {
    long i = (long)blockIdx.x * blockDim.x + threadIdx.x;
    if (i < n) out[i] = h0[i] + gacc[i] * 0.5f;
}
__global__ void combine_m(const float* __restrict__ h0, const float* __restrict__ s1,
                          const float* __restrict__ d1, const float* __restrict__ gacc,
                          float* __restrict__ out, int N) {
    long i = (long)blockIdx.x * blockDim.x + threadIdx.x;
    if (i >= (long)N * HD) return;
    const int n = (int)(i >> 7);
    const float dg = fmaxf(d1[n], 1.0f);
    out[i] = h0[i] + (s1[i] / dg + gacc[i]) * (1.0f / 3.0f);
}
__global__ void combine_r(const float* __restrict__ h0, const float* __restrict__ s1,
                          const float* __restrict__ d1, const float* __restrict__ s2,
                          const float* __restrict__ d2, const float* __restrict__ gacc,
                          float* __restrict__ out, int N) {
    long i = (long)blockIdx.x * blockDim.x + threadIdx.x;
    if (i >= (long)N * HD) return;
    const int n = (int)(i >> 7);
    const float g1 = fmaxf(d1[n], 1.0f), g2 = fmaxf(d2[n], 1.0f);
    out[i] = h0[i] + (s1[i] / g1 + s2[i] / g2 + gacc[i]) * (1.0f / 3.0f);
}

// =======================================================================
extern "C" void kernel_launch(void* const* d_in, const int* in_sizes, int n_in,
                              void* d_out, int out_size, void* d_ws, size_t ws_size,
                              hipStream_t stream) {
    const float* h_enz = (const float*)d_in[0];
    const float* h_met = (const float*)d_in[1];
    const float* h_rxn = (const float*)d_in[2];
    const float* ef_mod = (const float*)d_in[3];
    const float* ef_sig = (const float*)d_in[4];
    const float* ef_brg = (const float*)d_in[5];
    const float* ef_trn = (const float*)d_in[6];
    const float* ln1_g = (const float*)d_in[7];
    const float* ln1_b = (const float*)d_in[8];
    const float* ln2_g = (const float*)d_in[9];
    const float* ln2_b = (const float*)d_in[10];
    const float* wp = (const float*)d_in[11];
    const float* bp = (const float*)d_in[12];
    const float* wl = (const float*)d_in[13];
    const float* a_src = (const float*)d_in[14];
    const float* a_dst = (const float*)d_in[15];
    const float* we = (const float*)d_in[16];
    const float* a_edge = (const float*)d_in[17];
    const float* w1 = (const float*)d_in[18];
    const float* b1 = (const float*)d_in[19];
    const float* w2 = (const float*)d_in[20];
    const float* b2 = (const float*)d_in[21];
    const int* src_sub = (const int*)d_in[22];  const int* dst_sub = (const int*)d_in[23];
    const int* src_prod = (const int*)d_in[24]; const int* dst_prod = (const int*)d_in[25];
    const int* src_cat = (const int*)d_in[26];  const int* dst_cat = (const int*)d_in[27];
    const int* src_mod = (const int*)d_in[28];  const int* dst_mod = (const int*)d_in[29];
    const int* src_reg = (const int*)d_in[30];  const int* dst_reg = (const int*)d_in[31];
    const int* src_sig = (const int*)d_in[32];  const int* dst_sig = (const int*)d_in[33];
    const int* src_brg = (const int*)d_in[34];  const int* dst_brg = (const int*)d_in[35];
    const int* src_trn = (const int*)d_in[36];  const int* dst_trn = (const int*)d_in[37];
    float* out = (float*)d_out;

    const int NE = in_sizes[0] / HD, NM = in_sizes[1] / HD, NR = in_sizes[2] / HD;
    const int E = in_sizes[22];

    // ---------------- workspace arena ----------------
    char* base = (char*)d_ws;
    size_t cur = 0;
    auto alloc = [&](size_t n) -> void* {
        void* p = base + cur;
        cur = (cur + n + 255) & ~(size_t)255;
        return p;
    };
    _Float16* xe = (_Float16*)alloc((size_t)NE * HD * 2);
    _Float16* xm = (_Float16*)alloc((size_t)NM * HD * 2);
    _Float16* xr = (_Float16*)alloc((size_t)NR * HD * 2);
    _Float16* wph = (_Float16*)alloc(3 * HD * HD * 2);
    _Float16* wlh = (_Float16*)alloc(5 * HD * HD * 2);
    _Float16* w1h = (_Float16*)alloc(3 * HD * FFND * 2);
    _Float16* w2h = (_Float16*)alloc(3 * FFND * HD * 2);
    float* u = (float*)alloc(5 * 2 * 128 * 4 * 4);
    float* ce = (float*)alloc(128 * 4);
    const size_t markReuse = cur;  // everything below is dead by FFN time
    _Float16* Psub = (_Float16*)alloc((size_t)NM * HD * 2);
    _Float16* Pprod = (_Float16*)alloc((size_t)NR * HD * 2);
    _Float16* Pcat = (_Float16*)alloc((size_t)NE * HD * 2);
    _Float16* Zmod = (_Float16*)alloc((size_t)NE * HD * 2);
    _Float16* Zreg = (_Float16*)alloc((size_t)NE * HD * 2);
    _Float16* Zsig = (_Float16*)alloc((size_t)NM * HD * 2);
    _Float16* Zbrg = (_Float16*)alloc((size_t)NM * HD * 2);
    _Float16* Ztrn = (_Float16*)alloc((size_t)NM * HD * 2);
    float* ls_mod = (float*)alloc((size_t)NE * 16); float* ld_mod = (float*)alloc((size_t)NR * 16);
    float* ls_reg = (float*)alloc((size_t)NE * 16); float* ld_reg = (float*)alloc((size_t)NE * 16);
    float* ls_sig = (float*)alloc((size_t)NM * 16); float* ld_sig = (float*)alloc((size_t)NE * 16);
    float* ls_brg = (float*)alloc((size_t)NM * 16); float* ld_brg = (float*)alloc((size_t)NM * 16);
    float* ls_trn = (float*)alloc((size_t)NM * 16); float* ld_trn = (float*)alloc((size_t)NM * 16);
    float* elog = (float*)alloc((size_t)E * 16);
    unsigned* mx = (unsigned*)alloc((size_t)NM * 16);
    float* den = (float*)alloc((size_t)NM * 16);
    float* s_sub = (float*)alloc((size_t)NR * HD * 4);  float* deg_sub = (float*)alloc((size_t)NR * 4);
    float* s_cat = (float*)alloc((size_t)NR * HD * 4);  float* deg_cat = (float*)alloc((size_t)NR * 4);
    float* s_prod = (float*)alloc((size_t)NM * HD * 4); float* deg_prod = (float*)alloc((size_t)NM * 4);
    float* gacc_e = (float*)alloc((size_t)NE * HD * 4);
    float* gacc_m = (float*)alloc((size_t)NM * HD * 4);
    float* gacc_r = (float*)alloc((size_t)NR * HD * 4);
    _Float16* tbuf = (_Float16*)(base + markReuse);  // FFN scratch reuses msg-pass region

    // ---------------- phase 1: LN1 -> fp16, weight converts, u/ce ----------------
    ln_to_f16<<<NE / 8, 256, 0, stream>>>(h_enz, ln1_g + 0, ln1_b + 0, xe, NE);
    ln_to_f16<<<NM / 8, 256, 0, stream>>>(h_met, ln1_g + HD, ln1_b + HD, xm, NM);
    ln_to_f16<<<NR / 8, 256, 0, stream>>>(h_rxn, ln1_g + 2 * HD, ln1_b + 2 * HD, xr, NR);
    f32_to_f16<<<(3 * HD * HD + 255) / 256, 256, 0, stream>>>(wp, wph, 3 * HD * HD);
    f32_to_f16<<<(5 * HD * HD + 255) / 256, 256, 0, stream>>>(wl, wlh, 5 * HD * HD);
    f32_to_f16<<<(3 * HD * FFND + 255) / 256, 256, 0, stream>>>(w1, w1h, 3 * HD * FFND);
    f32_to_f16<<<(3 * FFND * HD + 255) / 256, 256, 0, stream>>>(w2, w2h, 3 * FFND * HD);
    prep_u<<<(5 * 2 * 128 * 4 + 255) / 256, 256, 0, stream>>>(wl, a_src, a_dst, u);
    prep_ce<<<1, 128, 0, stream>>>(we, a_edge, ce);

    // ---------------- phase 2: WMMA projections (fp16 out) ----------------
    auto G = [](int M) { return dim3(M / 128, 1); };
    gemm_wmma<3><<<G(NM), 256, 0, stream>>>(xm, HD, wph + 0 * HD * HD, HD, bp + 0, nullptr, Psub, nullptr, HD, HD);
    gemm_wmma<3><<<G(NR), 256, 0, stream>>>(xr, HD, wph + 1 * HD * HD, HD, bp + HD, nullptr, Pprod, nullptr, HD, HD);
    gemm_wmma<3><<<G(NE), 256, 0, stream>>>(xe, HD, wph + 2 * HD * HD, HD, bp + 2 * HD, nullptr, Pcat, nullptr, HD, HD);
    gemm_wmma<3><<<G(NE), 256, 0, stream>>>(xe, HD, wlh + 0 * HD * HD, HD, nullptr, nullptr, Zmod, nullptr, HD, HD);
    gemm_wmma<3><<<G(NE), 256, 0, stream>>>(xe, HD, wlh + 1 * HD * HD, HD, nullptr, nullptr, Zreg, nullptr, HD, HD);
    gemm_wmma<3><<<G(NM), 256, 0, stream>>>(xm, HD, wlh + 2 * HD * HD, HD, nullptr, nullptr, Zsig, nullptr, HD, HD);
    gemm_wmma<3><<<G(NM), 256, 0, stream>>>(xm, HD, wlh + 3 * HD * HD, HD, nullptr, nullptr, Zbrg, nullptr, HD, HD);
    gemm_wmma<3><<<G(NM), 256, 0, stream>>>(xm, HD, wlh + 4 * HD * HD, HD, nullptr, nullptr, Ztrn, nullptr, HD, HD);

    // ---------------- phase 3: node attention logits ----------------
    auto ND = [&](const _Float16* x, int r, int sd, float* o, int N) {
        node_dot<<<(N + 255) / 256, 256, 0, stream>>>(x, u + (r * 2 + sd) * 512, o, N);
    };
    ND(xe, 0, 0, ls_mod, NE); ND(xr, 0, 1, ld_mod, NR);
    ND(xe, 1, 0, ls_reg, NE); ND(xe, 1, 1, ld_reg, NE);
    ND(xm, 2, 0, ls_sig, NM); ND(xe, 2, 1, ld_sig, NE);
    ND(xm, 3, 0, ls_brg, NM); ND(xm, 3, 1, ld_brg, NM);
    ND(xm, 4, 0, ls_trn, NM); ND(xm, 4, 1, ld_trn, NM);

    // ---------------- phase 4: zero accumulators ----------------
    (void)hipMemsetAsync(s_sub, 0, (size_t)NR * HD * 4, stream);
    (void)hipMemsetAsync(deg_sub, 0, (size_t)NR * 4, stream);
    (void)hipMemsetAsync(s_cat, 0, (size_t)NR * HD * 4, stream);
    (void)hipMemsetAsync(deg_cat, 0, (size_t)NR * 4, stream);
    (void)hipMemsetAsync(s_prod, 0, (size_t)NM * HD * 4, stream);
    (void)hipMemsetAsync(deg_prod, 0, (size_t)NM * 4, stream);
    (void)hipMemsetAsync(gacc_e, 0, (size_t)NE * HD * 4, stream);
    (void)hipMemsetAsync(gacc_m, 0, (size_t)NM * HD * 4, stream);
    (void)hipMemsetAsync(gacc_r, 0, (size_t)NR * HD * 4, stream);

    // ---------------- phase 5: phys scatter ----------------
    const int EB = (E * 32) / 256;
    phys_scatter<<<EB, 256, 0, stream>>>(Psub, src_sub, dst_sub, s_sub, deg_sub, E);
    phys_scatter<<<EB, 256, 0, stream>>>(Pprod, src_prod, dst_prod, s_prod, deg_prod, E);
    phys_scatter<<<EB, 256, 0, stream>>>(Pcat, src_cat, dst_cat, s_cat, deg_cat, E);

    // ---------------- phase 6: GAT relations (sequential, shared scratch) ----------------
    auto GAT = [&](const float* ls, const float* ld, const float* ef, const float* cer,
                   const int* s, const int* d, const _Float16* Z, float* acc, int Nd) {
        (void)hipMemsetAsync(mx, 0, (size_t)Nd * 16, stream);
        (void)hipMemsetAsync(den, 0, (size_t)Nd * 16, stream);
        gat_logit<<<(E + 255) / 256, 256, 0, stream>>>(ls, ld, ef, cer, s, d, elog, mx, E);
        gat_exp<<<(E + 255) / 256, 256, 0, stream>>>(elog, mx, den, d, E);
        gat_scatter<<<EB, 256, 0, stream>>>(Z, elog, den, s, d, acc, E);
    };
    GAT(ls_mod, ld_mod, ef_mod, ce + 0 * 32, src_mod, dst_mod, Zmod, gacc_r, NR);
    GAT(ls_reg, ld_reg, nullptr, nullptr,    src_reg, dst_reg, Zreg, gacc_e, NE);
    GAT(ls_sig, ld_sig, ef_sig, ce + 1 * 32, src_sig, dst_sig, Zsig, gacc_e, NE);
    GAT(ls_brg, ld_brg, ef_brg, ce + 2 * 32, src_brg, dst_brg, Zbrg, gacc_m, NM);
    GAT(ls_trn, ld_trn, ef_trn, ce + 3 * 32, src_trn, dst_trn, Ztrn, gacc_m, NM);

    // ---------------- phase 7: residual combine -> d_out ----------------
    float* out_e = out;
    float* out_m = out + (size_t)NE * HD;
    float* out_r = out + (size_t)(NE + NM) * HD;
    combine_e<<<((long)NE * HD + 255) / 256, 256, 0, stream>>>(h_enz, gacc_e, out_e, (long)NE * HD);
    combine_m<<<((long)NM * HD + 255) / 256, 256, 0, stream>>>(h_met, s_prod, deg_prod, gacc_m, out_m, NM);
    combine_r<<<((long)NR * HD + 255) / 256, 256, 0, stream>>>(h_rxn, s_sub, deg_sub, s_cat, deg_cat, gacc_r, out_r, NR);

    // ---------------- phase 8: FFN per node type (LN2 -> GEMM+GELU -> GEMM+residual) ----------------
    float* outs[3] = {out_e, out_m, out_r};
    int Ns[3] = {NE, NM, NR};
    _Float16* ys[3] = {xe, xm, xr};  // reuse LN1 fp16 buffers
    for (int t = 0; t < 3; ++t) {
        const int N = Ns[t];
        ln_to_f16<<<N / 8, 256, 0, stream>>>(outs[t], ln2_g + t * HD, ln2_b + t * HD, ys[t], N);
        gemm_wmma<1><<<dim3(N / 128, FFND / 128), 256, 0, stream>>>(
            ys[t], HD, w1h + (size_t)t * HD * FFND, FFND, b1 + t * FFND,
            nullptr, tbuf, nullptr, FFND, HD);
        gemm_wmma<2><<<dim3(N / 128, 1), 256, 0, stream>>>(
            tbuf, FFND, w2h + (size_t)t * FFND * HD, HD, b2 + t * HD,
            outs[t], nullptr, outs[t], HD, FFND);
    }
}
